// CasualMultiQueryAttention_53386443489946
// MI455X (gfx1250) — matrix operand order repaired and checked by reading, not compiled, not run on hardware
//
#include <hip/hip_runtime.h>

// ---------------------------------------------------------------------------
// Problem constants (B=2, S=2048, D=2048, H=16, HD=128)
// ---------------------------------------------------------------------------
static constexpr int BB  = 2;
static constexpr int SS  = 2048;
static constexpr int DDim = 2048;
static constexpr int HH  = 16;
static constexpr int HDm = 128;          // head dim
static constexpr int NT  = BB * SS;      // 4096 token rows

typedef __attribute__((ext_vector_type(16))) __bf16 v16bf;
typedef __attribute__((ext_vector_type(8)))  float  v8f;

#define WMMA_BF16(Af, Bf, Cf) \
  __builtin_amdgcn_wmma_f32_16x16x32_bf16(false, (Af), false, (Bf), (short)0, (Cf), false, false)

// ---------------------------------------------------------------------------
// Fragment loaders (wave32, v_wmma_f32_16x16x32_bf16 layouts per CDNA5 ISA)
// A (16x32, M x K): lane m = lane&15; lanes<16: K in {0..7,16..23},
//                   lanes>=16: K in {8..15,24..31}; K-pairs contiguous.
// B (32x16, K x N): lane n = lane&15; K in {8h..8h+7, 16+8h..16+8h+7}, h=lane>>4
//                   -> loads from an n-major [N][K] source (2 x b128).
// C/D (16x16 f32):  n = lane&15, m = reg + 8*(lane>>4).
// ---------------------------------------------------------------------------
__device__ __forceinline__ v16bf load_a_frag(const __bf16* base, int stride, int lane) {
  const int m  = lane & 15;
  const int hf = (lane >> 4) & 1;
  union { v16bf v; unsigned u[8]; } f;
  const __bf16* row = base + (size_t)m * stride + 8 * hf;
#pragma unroll
  for (int p = 0; p < 8; ++p) {
    const int kk = (p < 4) ? (2 * p) : (16 + 2 * (p - 4));
    f.u[p] = *(const unsigned*)(row + kk);
  }
  return f.v;
}

__device__ __forceinline__ v16bf load_b_frag(const __bf16* base, int stride, int lane) {
  const int n  = lane & 15;
  const int hf = (lane >> 4) & 1;
  union { v16bf v; uint4 q[2]; } f;
  const __bf16* row = base + (size_t)n * stride + 8 * hf;
  f.q[0] = *(const uint4*)(row);
  f.q[1] = *(const uint4*)(row + 16);
  return f.v;
}

// ---------------------------------------------------------------------------
// Kernel 1: fp32 -> bf16 conversion (vectorized x4)
// ---------------------------------------------------------------------------
__global__ void cvt_f32_bf16_kernel(const float* __restrict__ in,
                                    __bf16* __restrict__ out, int n4) {
  int i = blockIdx.x * blockDim.x + threadIdx.x;
  if (i >= n4) return;
  float4 f = ((const float4*)in)[i];
  union { __bf16 b[4]; uint2 u; } o;
  o.b[0] = (__bf16)f.x; o.b[1] = (__bf16)f.y;
  o.b[2] = (__bf16)f.z; o.b[3] = (__bf16)f.w;
  ((uint2*)out)[i] = o.u;
}

// ---------------------------------------------------------------------------
// Kernel 2/5: C[M x N] = A[M x K](bf16) @ W[N x K]^T (bf16) + bias, out bf16/f32
// Block: 256 thr = 8 waves, tile BM=128 x BN=64, BK=32.
// Waves 4x2, each wave owns 32x32 = 2x2 WMMA tiles.
// ---------------------------------------------------------------------------
template <bool OUT_F32>
__global__ void __launch_bounds__(256)
gemm_bias_kernel(const __bf16* __restrict__ A, const __bf16* __restrict__ W,
                 const float* __restrict__ bias, void* __restrict__ Cout,
                 int M, int Ncols, int K) {
  __shared__ __bf16 As[128][32];   // row-major [m][k]   (8 KB)
  __shared__ __bf16 Ws[64][32];    // row-major [n][k]   (4 KB) == B n-major
  const int tid  = threadIdx.x;
  const int lane = tid & 31;
  const int wave = tid >> 5;
  const int wm = wave >> 1;        // 0..3
  const int wn = wave & 1;         // 0..1
  const int m0 = blockIdx.y * 128;
  const int n0 = blockIdx.x * 64;

  v8f acc[2][2] = {};

  const int ra = tid >> 1, ca = (tid & 1) * 16;     // A tile cooperative load
  const int rw = tid >> 2, cw = (tid & 3) * 8;      // W tile cooperative load

  for (int kb = 0; kb < K; kb += 32) {
    const __bf16* asrc = A + (size_t)(m0 + ra) * K + kb + ca;
    const __bf16* wsrc = W + (size_t)(n0 + rw) * K + kb + cw;
    *(uint4*)&As[ra][ca]     = *(const uint4*)(asrc);
    *(uint4*)&As[ra][ca + 8] = *(const uint4*)(asrc + 8);
    *(uint4*)&Ws[rw][cw]     = *(const uint4*)(wsrc);
    if (kb + 32 < K) {                       // prefetch next K slab
      __builtin_prefetch(asrc + 32, 0, 0);
      __builtin_prefetch(wsrc + 32, 0, 0);
    }
    __syncthreads();

    v16bf a0 = load_a_frag(&As[wm * 32][0],      32, lane);
    v16bf a1 = load_a_frag(&As[wm * 32 + 16][0], 32, lane);
    v16bf b0 = load_b_frag(&Ws[wn * 32][0],      32, lane);
    v16bf b1 = load_b_frag(&Ws[wn * 32 + 16][0], 32, lane);

    acc[0][0] = WMMA_BF16(a0, b0, acc[0][0]);
    acc[0][1] = WMMA_BF16(a0, b1, acc[0][1]);
    acc[1][0] = WMMA_BF16(a1, b0, acc[1][0]);
    acc[1][1] = WMMA_BF16(a1, b1, acc[1][1]);
    __syncthreads();
  }

  const int hf = lane >> 4, l16 = lane & 15;
#pragma unroll
  for (int i = 0; i < 2; ++i)
#pragma unroll
    for (int j = 0; j < 2; ++j)
#pragma unroll
      for (int r = 0; r < 8; ++r) {
        const int gm = m0 + wm * 32 + i * 16 + r + 8 * hf;
        const int gn = n0 + wn * 32 + j * 16 + l16;
        const float vout = acc[i][j][r] + bias[gn];
        if (OUT_F32) ((float*)Cout)[(size_t)gm * Ncols + gn] = vout;
        else         ((__bf16*)Cout)[(size_t)gm * Ncols + gn] = (__bf16)vout;
      }
}

// ---------------------------------------------------------------------------
// Kernel 3: RoPE in-place on bf16 Q and K. One thread per (token, head, pair).
// ---------------------------------------------------------------------------
__global__ void rope_kernel(__bf16* __restrict__ q, __bf16* __restrict__ k,
                            const float* __restrict__ fc,
                            const float* __restrict__ fs, int total) {
  int idx = blockIdx.x * blockDim.x + threadIdx.x;
  if (idx >= total) return;
  const int i = idx & 63;              // pair index within head dim
  const int t = idx >> 6;
  const int h = t & (HH - 1);
  const int n = t >> 4;                // token row (b*S + s)
  const int s = n & (SS - 1);
  const float c  = fc[s * (HDm / 2) + i];
  const float sn = fs[s * (HDm / 2) + i];
  const size_t off = (size_t)n * DDim + h * HDm + 2 * i;

  union { unsigned u; __bf16 b[2]; } pq, pk;
  pq.u = *(unsigned*)(q + off);
  pk.u = *(unsigned*)(k + off);
  {
    const float t1 = (float)pq.b[0], t2 = (float)pq.b[1];
    pq.b[0] = (__bf16)(t1 * c - t2 * sn);
    pq.b[1] = (__bf16)(t1 * sn + t2 * c);
  }
  {
    const float t1 = (float)pk.b[0], t2 = (float)pk.b[1];
    pk.b[0] = (__bf16)(t1 * c - t2 * sn);
    pk.b[1] = (__bf16)(t1 * sn + t2 * c);
  }
  *(unsigned*)(q + off) = pq.u;
  *(unsigned*)(k + off) = pk.u;
}

// ---------------------------------------------------------------------------
// Kernel 4: causal flash attention.
// Grid: (S/128, H, B); block 256 = 8 waves. Wave w owns q-rows [q0b+16w, +16).
// Loop over 32-key blocks (uniform trip count; causality via masking).
//   QK^T: B-frag reads Ks[key][hd] row-major directly.
//   P*V : B-frag needs keys contiguous per hd column -> Vt[hd][key] transposed.
// ---------------------------------------------------------------------------
__global__ void __launch_bounds__(256)
attn_kernel(const __bf16* __restrict__ q, const __bf16* __restrict__ k,
            const __bf16* __restrict__ v, __bf16* __restrict__ o) {
  __shared__ __bf16 Ks[32][HDm];       // key-major K tile   (8 KB)
  __shared__ __bf16 Vt[HDm][32];       // hd-major  V tile   (8 KB)
  __shared__ __bf16 Pt[8][16][32];     // per-wave P tile    (8 KB)

  const int tid  = threadIdx.x;
  const int lane = tid & 31;
  const int wave = tid >> 5;
  const int hf   = lane >> 4;
  const int l16  = lane & 15;
  const int q0b  = blockIdx.x * 128;
  const int h    = blockIdx.y;
  const int b    = blockIdx.z;
  const int q0   = q0b + wave * 16;
  const float scale = 0.08838834764831845f;   // 1/sqrt(128)

  // Q A-fragments, resident for the whole loop (16 rows x 128 hd)
  const __bf16* qbase = q + ((size_t)(b * SS + q0) * DDim + h * HDm);
  v16bf qa[4];
#pragma unroll
  for (int f = 0; f < 4; ++f) qa[f] = load_a_frag(qbase + f * 32, DDim, lane);

  v8f   Oacc[8] = {};
  float om[8], ol[8];
#pragma unroll
  for (int r = 0; r < 8; ++r) { om[r] = -1e30f; ol[r] = 0.f; }

  const int key_ld = tid >> 3;            // 0..31
  const int c_ld   = (tid & 7) * 16;      // 0..112
  const int jn = q0b / 32 + 4;            // cover keys 0 .. q0b+127

  for (int j = 0; j < jn; ++j) {
    const int k0 = j * 32;
    // cooperative staging of K (row-major) and V (transposed)
    {
      const size_t row = (size_t)(b * SS + k0 + key_ld) * DDim + h * HDm + c_ld;
      const __bf16* ksrc = k + row;
      *(uint4*)&Ks[key_ld][c_ld]     = *(const uint4*)(ksrc);
      *(uint4*)&Ks[key_ld][c_ld + 8] = *(const uint4*)(ksrc + 8);
      const __bf16* vsrc = v + row;
      __bf16 tmp[16];
      *(uint4*)(tmp)     = *(const uint4*)(vsrc);
      *(uint4*)(tmp + 8) = *(const uint4*)(vsrc + 8);
#pragma unroll
      for (int i = 0; i < 16; ++i) Vt[c_ld + i][key_ld] = tmp[i];
    }
    __syncthreads();

    // scores: S[16 x 32] as two 16x16 fragments, hd reduced in 4 WMMA steps
    v8f s0 = {}, s1 = {};
#pragma unroll
    for (int f = 0; f < 4; ++f) {
      v16bf kb0 = load_b_frag(&Ks[0][f * 32],  HDm, lane);
      v16bf kb1 = load_b_frag(&Ks[16][f * 32], HDm, lane);
      s0 = WMMA_BF16(qa[f], kb0, s0);
      s1 = WMMA_BF16(qa[f], kb1, s1);
    }

    // online softmax (row reductions across 16 lanes of the half-wave)
#pragma unroll
    for (int r = 0; r < 8; ++r) {
      const int m    = r + 8 * hf;
      const int qrow = q0 + m;
      const float v0 = (k0 + l16      <= qrow) ? s0[r] * scale : -1e30f;
      const float v1 = (k0 + 16 + l16 <= qrow) ? s1[r] * scale : -1e30f;
      float mx = fmaxf(v0, v1);
#pragma unroll
      for (int off = 1; off < 16; off <<= 1)
        mx = fmaxf(mx, __shfl_xor(mx, off, 32));
      const float mnew = fmaxf(om[r], mx);
      const float corr = __expf(om[r] - mnew);
      const float e0 = __expf(v0 - mnew);
      const float e1 = __expf(v1 - mnew);
      float ps = e0 + e1;
#pragma unroll
      for (int off = 1; off < 16; off <<= 1)
        ps += __shfl_xor(ps, off, 32);
      ol[r] = ol[r] * corr + ps;
      om[r] = mnew;
#pragma unroll
      for (int t = 0; t < 8; ++t) Oacc[t][r] *= corr;
      Pt[wave][m][l16]      = (__bf16)e0;
      Pt[wave][m][16 + l16] = (__bf16)e1;
    }

    // P * V: one A-frag (16x32 keys), 8 n-tiles over head dim
    v16bf pa = load_a_frag(&Pt[wave][0][0], 32, lane);
#pragma unroll
    for (int t = 0; t < 8; ++t) {
      v16bf vb = load_b_frag(&Vt[t * 16][0], 32, lane);
      Oacc[t] = WMMA_BF16(pa, vb, Oacc[t]);
    }
    __syncthreads();   // protect Ks/Vt before next block's staging
  }

  // normalize and store (bf16)
  __bf16* obase = o + ((size_t)(b * SS + q0) * DDim + h * HDm);
#pragma unroll
  for (int r = 0; r < 8; ++r) {
    const float inv = 1.0f / ol[r];
    const int m = r + 8 * hf;
#pragma unroll
    for (int t = 0; t < 8; ++t)
      obase[(size_t)m * DDim + t * 16 + l16] = (__bf16)(Oacc[t][r] * inv);
  }
}

// ---------------------------------------------------------------------------
// Host launcher. Inputs: x, wq, bq, wk, bk, wv, bv, wo, bo, freqs_cos, freqs_sin
// Workspace layout (bytes):
//   xb 0..16M | wqb 16M | wkb 24M | wvb 32M | wob 40M
//   qb 48M | kb 64M | vb 80M | ao 96M     (total 112 MB)
// ---------------------------------------------------------------------------
extern "C" void kernel_launch(void* const* d_in, const int* in_sizes, int n_in,
                              void* d_out, int out_size, void* d_ws, size_t ws_size,
                              hipStream_t stream) {
  (void)in_sizes; (void)n_in; (void)out_size; (void)ws_size;
  const float* x  = (const float*)d_in[0];
  const float* wq = (const float*)d_in[1];
  const float* bq = (const float*)d_in[2];
  const float* wk = (const float*)d_in[3];
  const float* bk = (const float*)d_in[4];
  const float* wv = (const float*)d_in[5];
  const float* bv = (const float*)d_in[6];
  const float* wo = (const float*)d_in[7];
  const float* bo = (const float*)d_in[8];
  const float* fc = (const float*)d_in[9];
  const float* fs = (const float*)d_in[10];

  char* ws = (char*)d_ws;
  const size_t MB = 1024u * 1024u;
  __bf16* xb  = (__bf16*)(ws + 0 * MB);
  __bf16* wqb = (__bf16*)(ws + 16 * MB);
  __bf16* wkb = (__bf16*)(ws + 24 * MB);
  __bf16* wvb = (__bf16*)(ws + 32 * MB);
  __bf16* wob = (__bf16*)(ws + 40 * MB);
  __bf16* qb  = (__bf16*)(ws + 48 * MB);
  __bf16* kb  = (__bf16*)(ws + 64 * MB);
  __bf16* vb  = (__bf16*)(ws + 80 * MB);
  __bf16* ao  = (__bf16*)(ws + 96 * MB);

  // 1) fp32 -> bf16
  const int nx4 = NT * DDim / 4;        // 2,097,152
  const int nw4 = DDim * DDim / 4;      // 1,048,576
  cvt_f32_bf16_kernel<<<(nx4 + 255) / 256, 256, 0, stream>>>(x,  xb,  nx4);
  cvt_f32_bf16_kernel<<<(nw4 + 255) / 256, 256, 0, stream>>>(wq, wqb, nw4);
  cvt_f32_bf16_kernel<<<(nw4 + 255) / 256, 256, 0, stream>>>(wk, wkb, nw4);
  cvt_f32_bf16_kernel<<<(nw4 + 255) / 256, 256, 0, stream>>>(wv, wvb, nw4);
  cvt_f32_bf16_kernel<<<(nw4 + 255) / 256, 256, 0, stream>>>(wo, wob, nw4);

  // 2) QKV projections (bf16 WMMA, fp32 accumulate, bf16 out)
  dim3 ggrid(DDim / 64, NT / 128);      // (32, 32)
  gemm_bias_kernel<false><<<ggrid, 256, 0, stream>>>(xb, wqb, bq, qb, NT, DDim, DDim);
  gemm_bias_kernel<false><<<ggrid, 256, 0, stream>>>(xb, wkb, bk, kb, NT, DDim, DDim);
  gemm_bias_kernel<false><<<ggrid, 256, 0, stream>>>(xb, wvb, bv, vb, NT, DDim, DDim);

  // 3) RoPE (in-place on qb, kb)
  const int nrope = NT * HH * (HDm / 2);   // 4,194,304
  rope_kernel<<<(nrope + 255) / 256, 256, 0, stream>>>(qb, kb, fc, fs, nrope);

  // 4) causal flash attention
  attn_kernel<<<dim3(SS / 128, HH, BB), 256, 0, stream>>>(qb, kb, vb, ao);

  // 5) output projection -> fp32 d_out
  gemm_bias_kernel<true><<<ggrid, 256, 0, stream>>>(ao, wob, bo, (float*)d_out, NT, DDim, DDim);
}
